// DetectionLayer_43104291782862
// MI455X (gfx1250) — compile-verified, matrix operand order checked by reading, not run on hardware
//
#include <hip/hip_runtime.h>
#include <hip/hip_bf16.h>

// ---------------- problem constants (match reference) ----------------
#define B_      8
#define N_      2000
#define C_      81
#define NPAD    2048          // padded N for WMMA tiling (multiple of 32)
#define MAXDET  100
#define META_STRIDE (12 + C_) // 93 floats per image_meta row
#define MIN_CONF 0.7f
#define NMS_THR  0.3f
#define CLASS_OFF 2.0f

typedef __attribute__((ext_vector_type(16))) _Float16 v16h;
typedef __attribute__((ext_vector_type(8)))  float    v8f;

// ---------------------------------------------------------------------
// Kernel 1: per-ROI class argmax, delta refinement, clipping, validity.
// ---------------------------------------------------------------------
__global__ __launch_bounds__(256)
void refine_kernel(const float* __restrict__ rois,
                   const float* __restrict__ probs,
                   const float* __restrict__ deltas,
                   const float* __restrict__ meta,
                   float* __restrict__ box,    // B*N*4
                   float* __restrict__ cls,    // B*N
                   float* __restrict__ score,  // B*N
                   float* __restrict__ key)    // B*N  (valid ? score : -1)
{
    int gid = blockIdx.x * blockDim.x + threadIdx.x;
    if (gid >= B_ * N_) return;
    int b = gid / N_;

    // window = (meta[b,7:11] - [0,0,1,1]) / ([h,w,h,w]-1), h/w from meta row 0
    float h  = meta[4];
    float w  = meta[5];
    float sy = h - 1.0f, sx = w - 1.0f;
    const float* mb = meta + (size_t)b * META_STRIDE;
    float wy1 = mb[7]  / sy;
    float wx1 = mb[8]  / sx;
    float wy2 = (mb[9]  - 1.0f) / sy;
    float wx2 = (mb[10] - 1.0f) / sx;

    const float* p = probs + (size_t)gid * C_;
    __builtin_prefetch(p, 0, 0);                 // global_prefetch_b8
    int best_c = 0;
    float best = p[0];
    for (int c = 1; c < C_; ++c) {
        float v = p[c];
        if (v > best) { best = v; best_c = c; }  // first-max semantics
    }

    const float* d = deltas + ((size_t)gid * C_ + best_c) * 4;
    float d0 = d[0] * 0.1f, d1 = d[1] * 0.1f;
    float d2 = d[2] * 0.2f, d3 = d[3] * 0.2f;

    const float* r = rois + (size_t)gid * 4;
    float y1 = r[0], x1 = r[1], y2 = r[2], x2 = r[3];
    float bh = y2 - y1, bw = x2 - x1;
    float cy = y1 + 0.5f * bh + d0 * bh;
    float cx = x1 + 0.5f * bw + d1 * bw;
    bh *= expf(d2);
    bw *= expf(d3);
    float ny1 = cy - 0.5f * bh;
    float nx1 = cx - 0.5f * bw;
    float ny2 = ny1 + bh;
    float nx2 = nx1 + bw;
    ny1 = fminf(fmaxf(ny1, wy1), wy2);
    nx1 = fminf(fmaxf(nx1, wx1), wx2);
    ny2 = fminf(fmaxf(ny2, wy1), wy2);
    nx2 = fminf(fmaxf(nx2, wx1), wx2);

    box[(size_t)gid * 4 + 0] = ny1;
    box[(size_t)gid * 4 + 1] = nx1;
    box[(size_t)gid * 4 + 2] = ny2;
    box[(size_t)gid * 4 + 3] = nx2;
    cls[gid]   = (float)best_c;
    score[gid] = best;
    bool valid = (best_c > 0) && (best >= MIN_CONF);
    key[gid]   = valid ? best : -1.0f;
}

// ---------------------------------------------------------------------
// Kernel 2: stable descending rank via WMMA (indicator-matrix x ones),
// then scatter into sorted arrays. One block (8 wave32) per image.
//
// rank[i] = sum_j ( key[j] > key[i]  ||  (key[j]==key[i] && j < i) )
// computed as D = A(16x32 f16 indicators) x B(ones) + C, chained over
// 64 K-chunks covering NPAD=2048 columns. Counts <= 2047 are exact
// (0/1 f16 operands, f32 accumulation).
// ---------------------------------------------------------------------
__global__ __launch_bounds__(256)
void rank_sort_kernel(const float* __restrict__ box,
                      const float* __restrict__ cls,
                      const float* __restrict__ score,
                      const float* __restrict__ key,
                      float* __restrict__ nmsbox_s,  // B*N*4 (box + class offset)
                      float* __restrict__ box_s,     // B*N*4 (plain refined box)
                      float* __restrict__ cls_s,     // B*N
                      float* __restrict__ score_s,   // B*N
                      float* __restrict__ keep_s)    // B*N (1 = initially kept)
{
    __shared__ float skey[NPAD];
    __shared__ int   srank[NPAD];

    int b   = blockIdx.x;
    int tid = threadIdx.x;
    const float* kb = key + (size_t)b * N_;

    for (int i = tid; i < NPAD; i += 256)
        skey[i] = (i < N_) ? kb[i] : -2.0f;     // pad keys below every real key
    __syncthreads();

    int lane = tid & 31;
    int wave = tid >> 5;
    int m    = lane & 15;                       // row within tile for this lane
    int koff = (lane >= 16) ? 8 : 0;            // K offset for upper half-wave

    v16h ones;
    #pragma unroll
    for (int e = 0; e < 16; ++e) ones[e] = (_Float16)1.0f;

    for (int t = wave; t < NPAD / 16; t += 8) { // uniform per wave -> EXEC all 1s
        int   i  = t * 16 + m;
        float ki = skey[i];
        v8f acc = {};
        for (int jc = 0; jc < NPAD / 32; ++jc) {
            v16h a;
            #pragma unroll
            for (int v = 0; v < 8; ++v) {
                int kbase = (v < 4) ? (koff + 2 * v)
                                    : (16 + koff + 2 * (v - 4));
                #pragma unroll
                for (int hh = 0; hh < 2; ++hh) {
                    int   j  = jc * 32 + kbase + hh;
                    float kj = skey[j];
                    bool ind = (kj > ki) || ((kj == ki) && (j < i));
                    a[2 * v + hh] = ind ? (_Float16)1.0f : (_Float16)0.0f;
                }
            }
            acc = __builtin_amdgcn_wmma_f32_16x16x32_f16(
                    false, a, false, ones, (short)0, acc, false, false);
        }
        // D layout: VGPR v -> (M=v, N=lane) for lanes 0-15; (M=v+8, N=lane-16)
        // for lanes 16-31. Read column N=0 at lanes 0 and 16.
        if (lane == 0 || lane == 16) {
            int rbase = t * 16 + ((lane == 16) ? 8 : 0);
            #pragma unroll
            for (int v = 0; v < 8; ++v) {
                int row = rbase + v;
                if (row < N_) srank[row] = (int)(acc[v] + 0.5f);
            }
        }
    }
    __syncthreads();

    // Scatter unsorted -> sorted position r = rank[i].
    for (int i = tid; i < N_; i += 256) {
        int    r   = srank[i];
        size_t src = (size_t)b * N_ + i;
        size_t dst = (size_t)b * N_ + r;
        float  c   = cls[src];
        float  off = c * CLASS_OFF;
        float b0 = box[src * 4 + 0], b1 = box[src * 4 + 1];
        float b2 = box[src * 4 + 2], b3 = box[src * 4 + 3];
        nmsbox_s[dst * 4 + 0] = b0 + off;
        nmsbox_s[dst * 4 + 1] = b1 + off;
        nmsbox_s[dst * 4 + 2] = b2 + off;
        nmsbox_s[dst * 4 + 3] = b3 + off;
        box_s[dst * 4 + 0] = b0;
        box_s[dst * 4 + 1] = b1;
        box_s[dst * 4 + 2] = b2;
        box_s[dst * 4 + 3] = b3;
        cls_s[dst]   = c;
        score_s[dst] = score[src];
        keep_s[dst]  = (key[src] >= 0.0f) ? 1.0f : 0.0f;
    }
}

// ---------------------------------------------------------------------
// Kernel 3: sequential NMS over sorted list (all state in LDS), then
// emit first MAXDET kept rows. One block per image.
// ---------------------------------------------------------------------
__global__ __launch_bounds__(256)
void nms_kernel(const float* __restrict__ nmsbox_s,
                const float* __restrict__ box_s,
                const float* __restrict__ cls_s,
                const float* __restrict__ score_s,
                const float* __restrict__ keep_s,
                float* __restrict__ out)       // B*MAXDET*6
{
    __shared__ float         sb[N_ * 4];
    __shared__ float         sarea[N_];
    __shared__ unsigned char skeep[N_];

    int b   = blockIdx.x;
    int tid = threadIdx.x;

    // Zero this image's output rows (harness poisons d_out).
    for (int i = tid; i < MAXDET * 6; i += 256)
        out[(size_t)b * MAXDET * 6 + i] = 0.0f;

    const float* nb = nmsbox_s + (size_t)b * N_ * 4;
    for (int i = tid; i < N_; i += 256) {
        float y1 = nb[i * 4 + 0], x1 = nb[i * 4 + 1];
        float y2 = nb[i * 4 + 2], x2 = nb[i * 4 + 3];
        sb[i * 4 + 0] = y1; sb[i * 4 + 1] = x1;
        sb[i * 4 + 2] = y2; sb[i * 4 + 3] = x2;
        sarea[i] = (y2 - y1) * (x2 - x1);
        skeep[i] = (keep_s[(size_t)b * N_ + i] > 0.5f) ? 1 : 0;
    }
    __syncthreads();

    for (int i = 0; i < N_ - 1; ++i) {
        if (skeep[i]) {
            float iy1 = sb[i * 4 + 0], ix1 = sb[i * 4 + 1];
            float iy2 = sb[i * 4 + 2], ix2 = sb[i * 4 + 3];
            float ia  = sarea[i];
            for (int j = i + 1 + tid; j < N_; j += 256) {
                float ih = fminf(iy2, sb[j * 4 + 2]) - fmaxf(iy1, sb[j * 4 + 0]);
                float iw = fminf(ix2, sb[j * 4 + 3]) - fmaxf(ix1, sb[j * 4 + 1]);
                ih = fmaxf(ih, 0.0f);
                iw = fmaxf(iw, 0.0f);
                float inter = ih * iw;
                float uni   = ia + sarea[j] - inter;
                float iou   = inter / (uni + 1e-8f);
                if (iou > NMS_THR) skeep[j] = 0;
            }
        }
        __syncthreads();
    }

    if (tid == 0) {
        int cnt = 0;
        for (int s = 0; s < N_ && cnt < MAXDET; ++s) {
            if (skeep[s]) {
                size_t src = (size_t)b * N_ + s;
                float* o   = out + ((size_t)b * MAXDET + cnt) * 6;
                o[0] = box_s[src * 4 + 0];
                o[1] = box_s[src * 4 + 1];
                o[2] = box_s[src * 4 + 2];
                o[3] = box_s[src * 4 + 3];
                o[4] = cls_s[src];
                o[5] = score_s[src];
                ++cnt;
            }
        }
    }
}

// ---------------------------------------------------------------------
extern "C" void kernel_launch(void* const* d_in, const int* in_sizes, int n_in,
                              void* d_out, int out_size, void* d_ws, size_t ws_size,
                              hipStream_t stream) {
    const float* rois   = (const float*)d_in[0];   // (8,2000,4)
    const float* probs  = (const float*)d_in[1];   // (8,2000,81)
    const float* deltas = (const float*)d_in[2];   // (8,2000,81,4)
    const float* meta   = (const float*)d_in[3];   // (8,93)

    float* ws       = (float*)d_ws;                // ~1.15 MB of floats
    float* box      = ws;                          // B*N*4
    float* cls      = box      + (size_t)B_ * N_ * 4;
    float* score    = cls      + (size_t)B_ * N_;
    float* key      = score    + (size_t)B_ * N_;
    float* nmsbox_s = key      + (size_t)B_ * N_;
    float* box_s    = nmsbox_s + (size_t)B_ * N_ * 4;
    float* cls_s    = box_s    + (size_t)B_ * N_ * 4;
    float* score_s  = cls_s    + (size_t)B_ * N_;
    float* keep_s   = score_s  + (size_t)B_ * N_;

    refine_kernel<<<(B_ * N_ + 255) / 256, 256, 0, stream>>>(
        rois, probs, deltas, meta, box, cls, score, key);
    rank_sort_kernel<<<B_, 256, 0, stream>>>(
        box, cls, score, key, nmsbox_s, box_s, cls_s, score_s, keep_s);
    nms_kernel<<<B_, 256, 0, stream>>>(
        nmsbox_s, box_s, cls_s, score_s, keep_s, (float*)d_out);
}
